// CausalSelfAttention_19765439496473
// MI455X (gfx1250) — matrix operand order repaired and checked by reading, not compiled, not run on hardware
//
#include <hip/hip_runtime.h>
#include <hip/hip_bf16.h>

// ---------------------------------------------------------------------------
// CDNA5 (gfx1250, wave32) causal self-attention:
//   1) QKV GEMM (bf16 WMMA, fp32 accum), epilogue scatters q/k/v [B,H,S,dh] bf16
//   2) Flash attention (bf16 WMMA QK^T and PV, fp32 online softmax,
//      TDM tensor_load_to_lds K-tile staging + global prefetch)
//   3) Output GEMM (bf16 WMMA) + bias -> fp32 d_out
// B=2, S=2048, D=1024, H=16, dh=64
// ---------------------------------------------------------------------------

typedef __attribute__((ext_vector_type(16))) __bf16 v16bf;
typedef __attribute__((ext_vector_type(8)))  float  v8f;
typedef __attribute__((ext_vector_type(4)))  unsigned int v4u;
typedef __attribute__((ext_vector_type(8)))  int   v8i_;
typedef __attribute__((ext_vector_type(4)))  int   v4i_;

union FragCvt { uint4 u[2]; v16bf v; };

__device__ __forceinline__ v8f wmma_bf16(v16bf a, v16bf b, v8f c) {
  return __builtin_amdgcn_wmma_f32_16x16x32_bf16(false, a, false, b, (short)0, c,
                                                 false, false);
}

__device__ __forceinline__ unsigned short f2bf(float f) {
  unsigned u = __float_as_uint(f);
  unsigned r = u + 0x7FFFu + ((u >> 16) & 1u);   // round-to-nearest-even
  return (unsigned short)(r >> 16);
}

#define B_   2
#define S_   2048
#define D_   1024
#define H_   16
#define DH_  64

#if defined(__gfx1250__) && __has_builtin(__builtin_amdgcn_tensor_load_to_lds)
#define ATTN_USE_TDM 1
#else
#define ATTN_USE_TDM 0
#endif

#if ATTN_USE_TDM
// TDM load of a 32(keys) x 64(dh) bf16 tile into LDS with padding so that the
// LDS row stride is 160B (64 bf16 data + 16 bf16 pad == KSTR below).
// D# layout per CDNA5 ISA (2D tensor, groups 2/3
// zero-filled; probe-verified that a zero-filled call compiles).
__device__ __forceinline__ void tdm_load_k_tile(unsigned lds_addr,
                                                const void* gptr) {
  unsigned long long ga = (unsigned long long)(size_t)gptr;
  v4u g0;
  g0.x = 0x1u;                                   // count=1, user descriptor
  g0.y = lds_addr;                               // LDS byte address
  g0.z = (unsigned)(ga & 0xFFFFFFFFu);           // global_addr[31:0]
  g0.w = (unsigned)((ga >> 32) & 0x01FFFFFFu)    // global_addr[56:32]
       | 0x80000000u;                            // type=2 ("image")
  v8i_ g1;
  g1[0] = (1 << 16)        // data_size = 2 bytes
        | (1 << 20)        // pad_enable (LDS dest padding, load only)
        | (4 << 22)        // pad_interval: 32 DWORDs (=128B data row)
        | (7 << 25);       // pad_amount : 8 DWORDs  (=32B pad -> 160B stride)
  g1[1] = (int)(64u << 16);    // tensor_dim0 = 64 (bits 79:48, low half)
  g1[2] = (int)(2048u << 16);  // tensor_dim0 hi | tensor_dim1 = 2048 (low 16)
  g1[3] = (int)(64u << 16);    // tensor_dim1 hi | tile_dim0 = 64
  g1[4] = 32;                  // tile_dim1 = 32 keys, tile_dim2 = 0
  g1[5] = 64;                  // tensor_dim0_stride = 64 elements
  g1[6] = 0;
  g1[7] = 0;
  v4i_ z4 = (v4i_)0;
#if __clang_major__ >= 23
  v8i_ z8 = (v8i_)0;
  __builtin_amdgcn_tensor_load_to_lds(g0, g1, z4, z4, z8, 0);
#else
  __builtin_amdgcn_tensor_load_to_lds(g0, g1, z4, z4, 0);
#endif
}
#endif

// ---------------------------------------------------------------------------
// Generic GEMM: C[M,N] = A[M,K](f32) * Bm[K,N](f32) + bias[N]
// Block tile 128x128, K-step 32. 8 waves: 4(M) x 2(N); wave tile 32x64.
// MODE 0: scatter to q/k/v bf16 buffers ([B,H,S,dh], interleaved 3*dh columns)
// MODE 1: plain fp32 store to out
// ---------------------------------------------------------------------------
#define ASTR 40   // 32 bf16 + 8 pad (80B row, 16B aligned)
#define BSTR 40

template <int MODE>
__global__ __launch_bounds__(256) void gemm_bf16_wmma(
    const float* __restrict__ A, const float* __restrict__ Bm,
    const float* __restrict__ bias, int K, int N,
    unsigned short* __restrict__ qb, unsigned short* __restrict__ kb,
    unsigned short* __restrict__ vb, float* __restrict__ out) {
  __shared__ __align__(16) unsigned short As[128 * ASTR];
  __shared__ __align__(16) unsigned short Bs[128 * BSTR];   // [n][k]

  const int t     = threadIdx.x;
  const int lane  = t & 31;
  const int wave  = t >> 5;
  const int wm    = wave & 3;        // 0..3  (M)
  const int wn    = wave >> 2;       // 0..1  (N)
  const int mBase = blockIdx.y * 128;
  const int nBase = blockIdx.x * 128;

  v8f acc[2][4];
#pragma unroll
  for (int mi = 0; mi < 2; ++mi)
#pragma unroll
    for (int ni = 0; ni < 4; ++ni) acc[mi][ni] = (v8f)0.0f;

  for (int ks = 0; ks < K; ks += 32) {
    __syncthreads();
    // stage A: 128 rows x 32 k  (f32 -> bf16)
#pragma unroll
    for (int p = 0; p < 4; ++p) {
      int lin = t + p * 256;            // 0..1023 float4s
      int row = lin >> 3;
      int c4  = lin & 7;
      float4 f = *(const float4*)&A[(size_t)(mBase + row) * K + ks + c4 * 4];
      ushort4 h;
      h.x = f2bf(f.x); h.y = f2bf(f.y); h.z = f2bf(f.z); h.w = f2bf(f.w);
      *(ushort4*)&As[row * ASTR + c4 * 4] = h;
    }
    // stage B transposed: Bs[n][k], 32 k-rows x 128 n-cols
#pragma unroll
    for (int p = 0; p < 4; ++p) {
      int lin = t + p * 256;            // 0..1023 float4s
      int kr  = lin >> 5;
      int c4  = lin & 31;
      float4 f = *(const float4*)&Bm[(size_t)(ks + kr) * N + nBase + c4 * 4];
      const float* ff = (const float*)&f;
#pragma unroll
      for (int j = 0; j < 4; ++j)
        Bs[(c4 * 4 + j) * BSTR + kr] = f2bf(ff[j]);
    }
    // prefetch next K-step tiles into cache while this step computes
    if (ks + 32 < K) {
      __builtin_prefetch(&A[(size_t)(mBase + (t >> 3)) * K + ks + 32], 0, 1);
      __builtin_prefetch(&Bm[(size_t)(ks + 32 + (t >> 3)) * N + nBase], 0, 1);
    }
    __syncthreads();

    // A fragments (2 M-subtiles)
    FragCvt fa[2];
    {
      int k0 = (lane < 16) ? 0 : 8;
#pragma unroll
      for (int mi = 0; mi < 2; ++mi) {
        int row = wm * 32 + mi * 16 + (lane & 15);
        fa[mi].u[0] = *(const uint4*)&As[row * ASTR + k0];
        fa[mi].u[1] = *(const uint4*)&As[row * ASTR + k0 + 16];
      }
    }
    // B fragments (4 N-subtiles)
    FragCvt fb[4];
    {
      int bk = (lane < 16) ? 0 : 8;
#pragma unroll
      for (int ni = 0; ni < 4; ++ni) {
        int n = wn * 64 + ni * 16 + (lane & 15);
        fb[ni].u[0] = *(const uint4*)&Bs[n * BSTR + bk];
        fb[ni].u[1] = *(const uint4*)&Bs[n * BSTR + bk + 16];
      }
    }
#pragma unroll
    for (int mi = 0; mi < 2; ++mi)
#pragma unroll
      for (int ni = 0; ni < 4; ++ni)
        acc[mi][ni] = wmma_bf16(fa[mi].v, fb[ni].v, acc[mi][ni]);
  }

  // epilogue  (C layout: VGPR r -> M = r + 8*(lane>=16); N = lane&15)
  const int half8 = (lane < 16) ? 0 : 8;
#pragma unroll
  for (int mi = 0; mi < 2; ++mi) {
#pragma unroll
    for (int ni = 0; ni < 4; ++ni) {
      int col = nBase + wn * 64 + ni * 16 + (lane & 15);
      float bv = bias[col];
#pragma unroll
      for (int r = 0; r < 8; ++r) {
        int   row = mBase + wm * 32 + mi * 16 + r + half8;
        float val = acc[mi][ni][r] + bv;
        if (MODE == 0) {
          int b = row >> 11;            // row / S_
          int s = row & (S_ - 1);
          int h = col / (3 * DH_);
          int rr = col - h * (3 * DH_);
          size_t base = (((size_t)b * H_ + h) * S_ + s) * DH_;
          unsigned short bf = f2bf(val);
          if (rr < DH_)            qb[base + rr]            = bf;
          else if (rr < 2 * DH_)   kb[base + rr - DH_]      = bf;
          else                     vb[base + rr - 2 * DH_]  = bf;
        } else {
          out[(size_t)row * N + col] = val;
        }
      }
    }
  }
}

// ---------------------------------------------------------------------------
// Flash attention: block = (b,h) x 128 query rows; 8 waves x 16 rows each.
// Key blocks of 32, causal-limited. bf16 WMMA, fp32 online softmax.
// K tile staged by the Tensor Data Mover (TDM) when available.
// ---------------------------------------------------------------------------
#define KSTR 80   // Ks[key][d]  : 64 bf16 + 16 pad (160B rows; matches TDM pad)
#define VSTR 40   // Vt[d][key]  : 32 bf16 + 8 pad  (80B rows)
#define PSTR 40   // Ps[row][key]: 32 bf16 + 8 pad

__global__ __launch_bounds__(256) void attn_flash_wmma(
    const unsigned short* __restrict__ qg, const unsigned short* __restrict__ kg,
    const unsigned short* __restrict__ vg, float* __restrict__ attnOut) {
  __shared__ __align__(16) unsigned short Ks[32 * KSTR];
  __shared__ __align__(16) unsigned short Vt[64 * VSTR];
  __shared__ __align__(16) unsigned short Ps[8 * 16 * PSTR];  // per-wave scratch

  const int lane  = threadIdx.x & 31;
  const int wave  = threadIdx.x >> 5;
  const int bh    = blockIdx.y;              // b*H + h
  const int qblk  = blockIdx.x * 128;
  const int half8 = (lane < 16) ? 0 : 8;
  const int qrow_base = qblk + wave * 16;
  const float SCALE = 0.125f;                // 1/sqrt(64)
  const float NEG_INF = -__builtin_inff();

  // Q fragments resident in registers (2 K-chunks of dh)
  FragCvt fq[2];
  {
    const unsigned short* qrow =
        qg + ((size_t)bh * S_ + qrow_base + (lane & 15)) * DH_;
#pragma unroll
    for (int c = 0; c < 2; ++c) {
      int k0 = c * 32 + ((lane < 16) ? 0 : 8);
      fq[c].u[0] = *(const uint4*)(qrow + k0);
      fq[c].u[1] = *(const uint4*)(qrow + k0 + 16);
    }
  }

  float m_i[8], l_i[8];
  v8f o[4];
#pragma unroll
  for (int r = 0; r < 8; ++r) { m_i[r] = NEG_INF; l_i[r] = 0.0f; }
#pragma unroll
  for (int i = 0; i < 4; ++i) o[i] = (v8f)0.0f;

  const int nkb = (qblk >> 5) + 4;           // causal: keys < qblk+128 only
  for (int kbi = 0; kbi < nkb; ++kbi) {
    const int key0 = kbi * 32;
    __syncthreads();                         // prior iteration done with LDS
#if ATTN_USE_TDM
    if (threadIdx.x < 32) {                  // wave 0 drives the DMA engine
      tdm_load_k_tile((unsigned)(size_t)(&Ks[0]),
                      kg + ((size_t)bh * S_ + key0) * DH_);
    }
#endif
    {   // stage V transposed (and K manually when no TDM): 32 keys x 64 dh
      int krow = threadIdx.x >> 3;           // 0..31
      int d0   = (threadIdx.x & 7) * 8;      // 0..56
      size_t gbase = ((size_t)bh * S_ + key0 + krow) * DH_ + d0;
#if !ATTN_USE_TDM
      uint4 kv = *(const uint4*)(kg + gbase);
      *(uint4*)&Ks[krow * KSTR + d0] = kv;
#endif
      uint4 vv = *(const uint4*)(vg + gbase);
      const unsigned short* pv = (const unsigned short*)&vv;
#pragma unroll
      for (int j = 0; j < 8; ++j) Vt[(d0 + j) * VSTR + krow] = pv[j];
      // prefetch next key block while this one is consumed
      if (kbi + 1 < nkb) {
        __builtin_prefetch(kg + gbase + 32 * DH_, 0, 1);
        __builtin_prefetch(vg + gbase + 32 * DH_, 0, 1);
      }
    }
#if ATTN_USE_TDM
    if (threadIdx.x < 32) __builtin_amdgcn_s_wait_tensorcnt(0);
#endif
    __syncthreads();

    // ---- scores: 16x32 = two 16x16 N-tiles, two dh K-chunks ----
    v8f s0 = (v8f)0.0f, s1 = (v8f)0.0f;
#pragma unroll
    for (int c = 0; c < 2; ++c) {
      int kd = c * 32 + ((lane < 16) ? 0 : 8);
      FragCvt fb0, fb1;
      int keyl = (lane & 15);
      fb0.u[0] = *(const uint4*)&Ks[keyl * KSTR + kd];
      fb0.u[1] = *(const uint4*)&Ks[keyl * KSTR + kd + 16];
      fb1.u[0] = *(const uint4*)&Ks[(keyl + 16) * KSTR + kd];
      fb1.u[1] = *(const uint4*)&Ks[(keyl + 16) * KSTR + kd + 16];
      s0 = wmma_bf16(fq[c].v, fb0.v, s0);
      s1 = wmma_bf16(fq[c].v, fb1.v, s1);
    }

    // ---- online softmax (per row: vgpr r, half -> M = r + half8) ----
#pragma unroll
    for (int r = 0; r < 8; ++r) {
      int   row = qrow_base + r + half8;
      int   ka  = key0 + (lane & 15);
      float sv0 = (ka      <= row) ? s0[r] * SCALE : NEG_INF;
      float sv1 = (ka + 16 <= row) ? s1[r] * SCALE : NEG_INF;
      float mx = fmaxf(sv0, sv1);
      mx = fmaxf(mx, __shfl_xor(mx, 1, 32));
      mx = fmaxf(mx, __shfl_xor(mx, 2, 32));
      mx = fmaxf(mx, __shfl_xor(mx, 4, 32));
      mx = fmaxf(mx, __shfl_xor(mx, 8, 32));
      float m_new  = fmaxf(m_i[r], mx);
      float factor = __expf(m_i[r] - m_new);
      float p0 = __expf(sv0 - m_new);
      float p1 = __expf(sv1 - m_new);
      float rs = p0 + p1;
      rs += __shfl_xor(rs, 1, 32);
      rs += __shfl_xor(rs, 2, 32);
      rs += __shfl_xor(rs, 4, 32);
      rs += __shfl_xor(rs, 8, 32);
      l_i[r] = l_i[r] * factor + rs;
      m_i[r] = m_new;
      o[0][r] *= factor; o[1][r] *= factor;
      o[2][r] *= factor; o[3][r] *= factor;
      int m = r + half8;
      Ps[(wave * 16 + m) * PSTR + (lane & 15)]      = f2bf(p0);
      Ps[(wave * 16 + m) * PSTR + 16 + (lane & 15)] = f2bf(p1);
    }

    // ---- O += P @ V  (LDS round-trip: C layout -> A layout; LDS is
    //      in-order per wave, so no barrier needed for wave-private Ps) ----
    FragCvt fp;
    {
      int prow = wave * 16 + (lane & 15);
      int pk0  = (lane < 16) ? 0 : 8;
      fp.u[0] = *(const uint4*)&Ps[prow * PSTR + pk0];
      fp.u[1] = *(const uint4*)&Ps[prow * PSTR + pk0 + 16];
    }
#pragma unroll
    for (int nt = 0; nt < 4; ++nt) {
      FragCvt fv;
      int dn = nt * 16 + (lane & 15);
      int vk = (lane < 16) ? 0 : 8;
      fv.u[0] = *(const uint4*)&Vt[dn * VSTR + vk];
      fv.u[1] = *(const uint4*)&Vt[dn * VSTR + vk + 16];
      o[nt] = wmma_bf16(fp.v, fv.v, o[nt]);
    }
  }

  // ---- finalize: O /= l, write [B,S,H*dh] fp32 ----
  const int b = bh >> 4, h = bh & 15;
#pragma unroll
  for (int r = 0; r < 8; ++r) {
    float inv  = 1.0f / l_i[r];
    int   srow = qrow_base + r + half8;
    size_t base = ((size_t)b * S_ + srow) * D_ + h * DH_;
#pragma unroll
    for (int nt = 0; nt < 4; ++nt)
      attnOut[base + nt * 16 + (lane & 15)] = o[nt][r] * inv;
  }
}

// ---------------------------------------------------------------------------
extern "C" void kernel_launch(void* const* d_in, const int* in_sizes, int n_in,
                              void* d_out, int out_size, void* d_ws, size_t ws_size,
                              hipStream_t stream) {
  const float* x    = (const float*)d_in[0];
  const float* Wqkv = (const float*)d_in[1];
  const float* bqkv = (const float*)d_in[2];
  const float* Wout = (const float*)d_in[3];
  const float* bout = (const float*)d_in[4];
  float* out = (float*)d_out;

  const size_t HEAD_ELEMS = (size_t)B_ * H_ * S_ * DH_;   // 4M elems
  unsigned short* qws = (unsigned short*)d_ws;
  unsigned short* kws = qws + HEAD_ELEMS;
  unsigned short* vws = kws + HEAD_ELEMS;
  float*          att = (float*)(vws + HEAD_ELEMS);

  dim3 blk(256);
  // 1) QKV projection -> per-head q/k/v (bf16)
  gemm_bf16_wmma<0><<<dim3((3 * D_) / 128, (B_ * S_) / 128), blk, 0, stream>>>(
      x, Wqkv, bqkv, D_, 3 * D_, qws, kws, vws, nullptr);
  // 2) causal flash attention -> att [B,S,D] fp32
  attn_flash_wmma<<<dim3(S_ / 128, B_ * H_), blk, 0, stream>>>(qws, kws, vws, att);
  // 3) output projection -> d_out
  gemm_bf16_wmma<1><<<dim3(D_ / 128, (B_ * S_) / 128), blk, 0, stream>>>(
      att, Wout, bout, D_, D_, nullptr, nullptr, nullptr, out);
}
